// ModelGNN_68899865362832
// MI455X (gfx1250) — matrix operand order repaired and checked by reading, not compiled, not run on hardware
//
#include <hip/hip_runtime.h>
#include <stdint.h>

// ---------------- problem constants ----------------
#define N_NODES  50000
#define N_GRAPHS 32
#define NODE_F   16
#define GLOBAL_F 2
#define HID      300
#define LAT      100
#define N_EDGES  800000

#define HIDP 320   // padded hidden: 20 N-tiles of 16, 10 K-chunks of 32
#define LATP 128   // padded latent: 8 N-tiles of 16
#define K0   32    // layer-0 edge input width (2*NODE_F) == one f16 WMMA K-chunk
#define K1   256   // layer-1 edge input width, 2*LAT=200 padded to 256

typedef _Float16 half_t;
typedef __attribute__((ext_vector_type(16))) _Float16 v16h;
typedef __attribute__((ext_vector_type(8)))  _Float16 v8h;
typedef __attribute__((ext_vector_type(8)))  float    v8f;

// CDNA5 split-counter LDS fence: our own wave's ds stores complete -> any lane
// may read any other lane's data.
__device__ __forceinline__ void wave_lds_fence() {
    asm volatile("s_wait_dscnt 0" ::: "memory");
}

// ---------------------------------------------------------------------------
// A-fragment loader (16x32 f16 tile, row-major in LDS, row stride `ld` halves).
// Per CDNA5 ISA 7.12.2: lane L<16 holds row M=L, K={0..7,16..23}; lane L>=16
// holds row M=L-16, K={8..15,24..31}  -> two contiguous 16-byte LDS reads.
// ---------------------------------------------------------------------------
__device__ __forceinline__ v16h load_a_frag(const half_t* tile, int ld, int kt, int lane) {
    int m    = lane & 15;
    int off8 = (lane >> 4) << 3;             // 0 or 8
    const half_t* p = tile + m * ld + kt * 32 + off8;
    v8h lo = *(const v8h*)(p);
    v8h hi = *(const v8h*)(p + 16);
    v16h r;
#pragma unroll
    for (int i = 0; i < 8; ++i) { r[i] = lo[i]; r[i + 8] = hi[i]; }
    return r;
}

// ---------------------------------------------------------------------------
// Weight packer: fp32 W[kIn][nIn] (row-major) -> f16 B-fragment stream padded
// to KP x NP.  Fragment (kt,nt): lane holds column n=nt*16+(lane&15),
// K = kt*32 + (i>>3)*16 + (lane>>4)*8 + (i&7), 16 halves per lane (32B load).
// ---------------------------------------------------------------------------
__global__ void pack_weight(const float* __restrict__ W, half_t* __restrict__ P,
                            int kIn, int nIn, int KP, int NP) {
    int idx = blockIdx.x * blockDim.x + threadIdx.x;
    int total = KP * NP;
    if (idx >= total) return;
    int i    = idx & 15;
    int lane = (idx >> 4) & 31;
    int tile = idx >> 9;
    int ntn  = NP >> 4;
    int kt   = tile / ntn;
    int nt   = tile - kt * ntn;
    int n    = nt * 16 + (lane & 15);
    int k    = kt * 32 + ((i >> 3) << 4) + ((lane >> 4) << 3) + (i & 7);
    float v  = (k < kIn && n < nIn) ? W[k * nIn + n] : 0.0f;
    P[idx]   = (half_t)v;
}

// zero-pad a bias vector to NP floats -> branch-free epilogue reads
__global__ void pack_bias(const float* __restrict__ b, float* __restrict__ P,
                          int nIn, int NP) {
    int i = blockIdx.x * blockDim.x + threadIdx.x;
    if (i < NP) P[i] = (i < nIn) ? b[i] : 0.0f;
}

// ---------------------------------------------------------------------------
// GEMM stage: [16 x 32*KT] (LDS A) x [32*KT x 16*NT] (packed weights) with
// (padded) bias + ReLU epilogue, f16 result written to LDS H (row stride HIDP).
// ---------------------------------------------------------------------------
template <int KT, int NT>
__device__ __forceinline__ void gemm_stage(const half_t* __restrict__ Ain, int ldA,
                                           const half_t* __restrict__ Wp,
                                           const float* __restrict__ biasp,  // [16*NT] padded
                                           half_t* __restrict__ Hout, int lane) {
    v16h a[KT];
#pragma unroll
    for (int kt = 0; kt < KT; ++kt) a[kt] = load_a_frag(Ain, ldA, kt, lane);

#pragma unroll 1
    for (int nt = 0; nt < NT; ++nt) {
        // hide L2 latency of the next N-tile's B fragments
        if (nt + 1 < NT)
            __builtin_prefetch(Wp + (size_t)(nt + 1) * 512 + lane * 16, 0, 0);
        v8f acc = {};
#pragma unroll
        for (int kt = 0; kt < KT; ++kt) {
            v16h b = *(const v16h*)(Wp + (size_t)(kt * NT + nt) * 512 + lane * 16);
            acc = __builtin_amdgcn_wmma_f32_16x16x32_f16(false, a[kt], false, b,
                                                         (short)0, acc, false, false);
        }
        // C/D layout: lane<16 -> M=r,  N=lane ; lane>=16 -> M=8+r, N=lane-16
        int n    = (lane & 15) + nt * 16;
        float bv = biasp[n];                 // padded -> unconditional
        int mb   = (lane >> 4) << 3;
#pragma unroll
        for (int r = 0; r < 8; ++r) {
            float v = acc[r] + bv;
            v = v > 0.0f ? v : 0.0f;
            Hout[(mb + r) * HIDP + n] = (half_t)v;
        }
    }
}

// ---------------------------------------------------------------------------
// Fused per-edge MLP: build A tile (x_i, x_j - x_i) -> 3 GEMMs -> atomic
// scatter-add of messages into node accumulator S[N_NODES][LATP].
// One wave == 16 edges; 2 waves / block; waves fully independent.
// ---------------------------------------------------------------------------
template <int LAYER>
__global__ void __launch_bounds__(64)
edge_mlp(const float* __restrict__ x,      // LAYER==0: [N_NODES][16] f32
         const half_t* __restrict__ y,     // LAYER==1: [N_NODES][LATP] f16
         const int* __restrict__ src, const int* __restrict__ dst,
         const half_t* __restrict__ W1p, const float* __restrict__ b1p,
         const half_t* __restrict__ W2p, const float* __restrict__ b2p,
         const half_t* __restrict__ W3p, const float* __restrict__ b3p,
         float* __restrict__ S) {
    constexpr int KIN = (LAYER == 0) ? K0 : K1;
    constexpr int KT1 = KIN / 32;
    constexpr size_t PER_WAVE = (size_t)(16 * KIN + 2 * 16 * HIDP) * 2 + 64;

    extern __shared__ char smem_raw[];
    const int lane = threadIdx.x & 31;
    const int wave = threadIdx.x >> 5;

    char*   base = smem_raw + (size_t)wave * PER_WAVE;
    half_t* sA   = (half_t*)base;
    half_t* sH1  = sA + 16 * KIN;
    half_t* sH2  = sH1 + 16 * HIDP;
    int*    sDst = (int*)(sH2 + 16 * HIDP);

    const int tile   = blockIdx.x * 2 + wave;
    const int e_base = tile * 16;
    if (e_base >= N_EDGES) return;     // no block barriers used -> safe

    // ---- build A tile: [16 edges][KIN]  = concat(x_i, x_j - x_i), zero pad
    if (lane < 16) {
        int e = e_base + lane;
        sDst[lane] = (e < N_EDGES) ? dst[e] : 0;
    }
#pragma unroll 1
    for (int idx = lane; idx < 16 * KIN; idx += 32) {
        int r = idx / KIN;
        int k = idx - r * KIN;
        int e = e_base + r;
        float v = 0.0f;
        if (e < N_EDGES) {
            int d = dst[e], s = src[e];
            if (LAYER == 0) {
                if (k < NODE_F) v = x[d * NODE_F + k];
                else            v = x[s * NODE_F + (k - NODE_F)] - x[d * NODE_F + (k - NODE_F)];
            } else {
                if (k < LAT)          v = (float)y[(size_t)d * LATP + k];
                else if (k < 2 * LAT) {
                    int t = k - LAT;
                    v = (float)y[(size_t)s * LATP + t] - (float)y[(size_t)d * LATP + t];
                }
            }
        }
        sA[r * KIN + k] = (half_t)v;
    }
    wave_lds_fence();

    // ---- GEMM1: [16,KIN] x [KIN,320] -> sH1, ReLU
    gemm_stage<KT1, HIDP / 16>(sA, KIN, W1p, b1p, sH1, lane);
    wave_lds_fence();

    // ---- GEMM2: [16,320] x [320,320] -> sH2, ReLU
    gemm_stage<HIDP / 32, HIDP / 16>(sH1, HIDP, W2p, b2p, sH2, lane);
    wave_lds_fence();

    // ---- GEMM3: [16,320] x [320,128] -> messages, scatter-add to S[dst]
    {
        v16h a[HIDP / 32];
#pragma unroll
        for (int kt = 0; kt < HIDP / 32; ++kt) a[kt] = load_a_frag(sH2, HIDP, kt, lane);

#pragma unroll 1
        for (int nt = 0; nt < LATP / 16; ++nt) {
            if (nt + 1 < LATP / 16)
                __builtin_prefetch(W3p + (size_t)(nt + 1) * 512 + lane * 16, 0, 0);
            v8f acc = {};
#pragma unroll
            for (int kt = 0; kt < HIDP / 32; ++kt) {
                v16h b = *(const v16h*)(W3p + (size_t)(kt * (LATP / 16) + nt) * 512 + lane * 16);
                acc = __builtin_amdgcn_wmma_f32_16x16x32_f16(false, a[kt], false, b,
                                                             (short)0, acc, false, false);
            }
            int n = (lane & 15) + nt * 16;
            float bv = b3p[n];               // padded bias, unconditional
            if (n < LAT) {
                int mb = (lane >> 4) << 3;
#pragma unroll
                for (int r = 0; r < 8; ++r) {
                    int e = e_base + mb + r;
                    if (e < N_EDGES)
                        atomicAdd(&S[(size_t)sDst[mb + r] * LATP + n], acc[r] + bv);
                }
            }
        }
    }
}

// ---------------- small helper kernels ----------------
__global__ void count_index(const int* __restrict__ idxv, float* __restrict__ cnt, int n) {
    int i = blockIdx.x * blockDim.x + threadIdx.x;
    if (i < n) atomicAdd(&cnt[idxv[i]], 1.0f);
}

// scatter-mean finalize + ReLU, re-emit f16 node features (pad cols zeroed)
__global__ void finalize_nodes(const float* __restrict__ S, const float* __restrict__ cnt,
                               half_t* __restrict__ Y) {
    int idx = blockIdx.x * blockDim.x + threadIdx.x;
    if (idx >= N_NODES * LATP) return;
    int node = idx >> 7;
    int n    = idx & 127;
    float v  = 0.0f;
    if (n < LAT) {
        float c = cnt[node];
        if (c < 1.0f) c = 1.0f;
        v = S[idx] / c;
        if (v < 0.0f) v = 0.0f;
    }
    Y[idx] = (half_t)v;
}

__global__ void pool_feat(const half_t* __restrict__ Y, const int* __restrict__ batch,
                          float* __restrict__ pooled) {
    int idx = blockIdx.x * blockDim.x + threadIdx.x;
    if (idx >= N_NODES * LAT) return;
    int node = idx / LAT;
    int n    = idx - node * LAT;
    atomicAdd(&pooled[batch[node] * LAT + n], (float)Y[(size_t)node * LATP + n]);
}

// tiny 32-graph head MLP, fp32 scalar (negligible work)
__global__ void head_mlp(const float* __restrict__ pooled, const float* __restrict__ gcnt,
                         const float* __restrict__ u,
                         const float* __restrict__ w1, const float* __restrict__ bb1,
                         const float* __restrict__ w2, const float* __restrict__ bb2,
                         const float* __restrict__ w3, const float* __restrict__ bb3,
                         float* __restrict__ out) {
    int g = threadIdx.x;
    if (g >= N_GRAPHS) return;
    float in[LAT + GLOBAL_F];
    float c = gcnt[g];
    if (c < 1.0f) c = 1.0f;
    for (int i = 0; i < LAT; ++i) in[i] = pooled[g * LAT + i] / c;
    in[LAT]     = u[g * GLOBAL_F + 0];
    in[LAT + 1] = u[g * GLOBAL_F + 1];
    float h1[LAT];
    for (int n = 0; n < LAT; ++n) {
        float a = bb1[n];
        for (int k = 0; k < LAT + GLOBAL_F; ++k) a += in[k] * w1[k * LAT + n];
        h1[n] = a > 0.0f ? a : 0.0f;
    }
    float h2[LAT];
    for (int n = 0; n < LAT; ++n) {
        float a = bb2[n];
        for (int k = 0; k < LAT; ++k) a += h1[k] * w2[k * LAT + n];
        h2[n] = a > 0.0f ? a : 0.0f;
    }
    float o = bb3[0];
    for (int k = 0; k < LAT; ++k) o += h2[k] * w3[k];
    out[g] = o;
}

// ---------------------------------------------------------------------------
extern "C" void kernel_launch(void* const* d_in, const int* in_sizes, int n_in,
                              void* d_out, int out_size, void* d_ws, size_t ws_size,
                              hipStream_t stream) {
    (void)in_sizes; (void)n_in; (void)out_size; (void)ws_size;

    const float* x     = (const float*)d_in[0];
    const int*   eidx  = (const int*)d_in[1];
    const int*   batch = (const int*)d_in[2];
    const float* u     = (const float*)d_in[3];
    const float* l0_w1 = (const float*)d_in[4];  const float* l0_b1 = (const float*)d_in[5];
    const float* l0_w2 = (const float*)d_in[6];  const float* l0_b2 = (const float*)d_in[7];
    const float* l0_w3 = (const float*)d_in[8];  const float* l0_b3 = (const float*)d_in[9];
    const float* l1_w1 = (const float*)d_in[10]; const float* l1_b1 = (const float*)d_in[11];
    const float* l1_w2 = (const float*)d_in[12]; const float* l1_b2 = (const float*)d_in[13];
    const float* l1_w3 = (const float*)d_in[14]; const float* l1_b3 = (const float*)d_in[15];
    const float* lw1   = (const float*)d_in[16]; const float* lb1   = (const float*)d_in[17];
    const float* lw2   = (const float*)d_in[18]; const float* lb2   = (const float*)d_in[19];
    const float* lw3   = (const float*)d_in[20]; const float* lb3   = (const float*)d_in[21];

    const int* src = eidx;
    const int* dst = eidx + N_EDGES;

    // ---- carve workspace (all 256B aligned) ----
    uintptr_t p = (uintptr_t)d_ws;
    auto take = [&](size_t bytes) -> void* {
        void* r = (void*)p;
        p += (bytes + 255) & ~(size_t)255;
        return r;
    };
    half_t* W1p0 = (half_t*)take((size_t)K0 * HIDP * 2);
    half_t* W2p0 = (half_t*)take((size_t)HIDP * HIDP * 2);
    half_t* W3p0 = (half_t*)take((size_t)HIDP * LATP * 2);
    half_t* W1p1 = (half_t*)take((size_t)K1 * HIDP * 2);
    half_t* W2p1 = (half_t*)take((size_t)HIDP * HIDP * 2);
    half_t* W3p1 = (half_t*)take((size_t)HIDP * LATP * 2);
    float*  B1p0 = (float*)take(HIDP * 4);
    float*  B2p0 = (float*)take(HIDP * 4);
    float*  B3p0 = (float*)take(LATP * 4);
    float*  B1p1 = (float*)take(HIDP * 4);
    float*  B2p1 = (float*)take(HIDP * 4);
    float*  B3p1 = (float*)take(LATP * 4);
    float*  S    = (float*)take((size_t)N_NODES * LATP * 4);
    float*  cnt  = (float*)take((size_t)N_NODES * 4);
    half_t* Y    = (half_t*)take((size_t)N_NODES * LATP * 2);
    float*  pooled = (float*)take((size_t)N_GRAPHS * LAT * 4);
    float*  gcnt   = (float*)take((size_t)N_GRAPHS * 4);

    // ---- zero accumulators ----
    hipMemsetAsync(S, 0, (size_t)N_NODES * LATP * 4, stream);
    hipMemsetAsync(cnt, 0, (size_t)N_NODES * 4, stream);
    hipMemsetAsync(pooled, 0, (size_t)N_GRAPHS * LAT * 4, stream);
    hipMemsetAsync(gcnt, 0, (size_t)N_GRAPHS * 4, stream);

    // ---- pack weights (f16 fragment layout) + biases (zero-padded f32) ----
    auto packW = [&](const float* W, half_t* P, int kIn, int nIn, int KP, int NP) {
        int tot = KP * NP;
        pack_weight<<<(tot + 255) / 256, 256, 0, stream>>>(W, P, kIn, nIn, KP, NP);
    };
    auto packB = [&](const float* b, float* P, int nIn, int NP) {
        pack_bias<<<(NP + 255) / 256, 256, 0, stream>>>(b, P, nIn, NP);
    };
    packW(l0_w1, W1p0, 2 * NODE_F, HID, K0,   HIDP);
    packW(l0_w2, W2p0, HID,        HID, HIDP, HIDP);
    packW(l0_w3, W3p0, HID,        LAT, HIDP, LATP);
    packW(l1_w1, W1p1, 2 * LAT,    HID, K1,   HIDP);
    packW(l1_w2, W2p1, HID,        HID, HIDP, HIDP);
    packW(l1_w3, W3p1, HID,        LAT, HIDP, LATP);
    packB(l0_b1, B1p0, HID, HIDP);
    packB(l0_b2, B2p0, HID, HIDP);
    packB(l0_b3, B3p0, LAT, LATP);
    packB(l1_b1, B1p1, HID, HIDP);
    packB(l1_b2, B2p1, HID, HIDP);
    packB(l1_b3, B3p1, LAT, LATP);

    // ---- dst-degree (shared by both scatter-mean layers) ----
    count_index<<<(N_EDGES + 255) / 256, 256, 0, stream>>>(dst, cnt, N_EDGES);

    const int nblk  = (N_EDGES / 16 + 1) / 2;          // 2 waves (tiles) per block
    const size_t pw0 = (size_t)(16 * K0 + 2 * 16 * HIDP) * 2 + 64;
    const size_t pw1 = (size_t)(16 * K1 + 2 * 16 * HIDP) * 2 + 64;

    // ---- layer 0 ----
    edge_mlp<0><<<nblk, 64, 2 * pw0, stream>>>(x, nullptr, src, dst,
                                               W1p0, B1p0, W2p0, B2p0, W3p0, B3p0, S);
    finalize_nodes<<<(N_NODES * LATP + 255) / 256, 256, 0, stream>>>(S, cnt, Y);

    // ---- layer 1 ----
    hipMemsetAsync(S, 0, (size_t)N_NODES * LATP * 4, stream);
    edge_mlp<1><<<nblk, 64, 2 * pw1, stream>>>(nullptr, Y, src, dst,
                                               W1p1, B1p1, W2p1, B2p1, W3p1, B3p1, S);
    finalize_nodes<<<(N_NODES * LATP + 255) / 256, 256, 0, stream>>>(S, cnt, Y);

    // ---- mean pool + head ----
    pool_feat<<<(N_NODES * LAT + 255) / 256, 256, 0, stream>>>(Y, batch, pooled);
    count_index<<<(N_NODES + 255) / 256, 256, 0, stream>>>(batch, gcnt, N_NODES);
    head_mlp<<<1, 64, 0, stream>>>(pooled, gcnt, u, lw1, lb1, lw2, lb2, lw3, lb3,
                                   (float*)d_out);
}
